// CrossAttention_75136157876745
// MI455X (gfx1250) — matrix operand order repaired and checked by reading, not compiled, not run on hardware
//
#include <hip/hip_runtime.h>

// ---------------------------------------------------------------------------
// Cross-attention for MI455X (gfx1250, wave32, WMMA bf16 -> f32).
//   q = in1 @ Wq + bq ; k = in2 @ Wk + bk ; v = in2 @ Wv + bv
//   out = softmax(q k^T / sqrt(512)) @ v
// B=16, N1=N2=2048, D=512, L=256.
// ---------------------------------------------------------------------------

typedef __attribute__((ext_vector_type(16))) __bf16 v16bf;
typedef __attribute__((ext_vector_type(8)))  __bf16 v8bf;
typedef __attribute__((ext_vector_type(8)))  float  v8f;

#define WMMA_BF16(a, b, c) \
  __builtin_amdgcn_wmma_f32_16x16x32_bf16(false, (a), false, (b), (short)0, (c), false, false)

static constexpr int BATCH = 16;
static constexpr int N1 = 2048;
static constexpr int N2 = 2048;
static constexpr int DIM = 512;
static constexpr int LEN = 256;

__device__ __forceinline__ v8bf cvt8(const float* p) {
  float4 x = *reinterpret_cast<const float4*>(p);
  float4 y = *(reinterpret_cast<const float4*>(p) + 1);
  v8bf r;
  r[0] = (__bf16)x.x; r[1] = (__bf16)x.y; r[2] = (__bf16)x.z; r[3] = (__bf16)x.w;
  r[4] = (__bf16)y.x; r[5] = (__bf16)y.y; r[6] = (__bf16)y.z; r[7] = (__bf16)y.w;
  return r;
}

__device__ __forceinline__ v16bf cat16(v8bf lo, v8bf hi) {
  return __builtin_shufflevector(lo, hi, 0, 1, 2, 3, 4, 5, 6, 7,
                                 8, 9, 10, 11, 12, 13, 14, 15);
}

__device__ __forceinline__ float redmax16(float v) {
#pragma unroll
  for (int m = 1; m < 16; m <<= 1) v = fmaxf(v, __shfl_xor(v, m, 32));
  return v;
}
__device__ __forceinline__ float redsum16(float v) {
#pragma unroll
  for (int m = 1; m < 16; m <<= 1) v += __shfl_xor(v, m, 32);
  return v;
}

// --------------------------------------------------------------------------
// Kernel 0: W[K][N=512] fp32  ->  WT[N=512][K] bf16  (row-major transpose)
// --------------------------------------------------------------------------
__global__ void wt_transpose_kernel(const float* __restrict__ W,
                                    __bf16* __restrict__ WT, int K) {
  int idx = blockIdx.x * 256 + threadIdx.x;
  if (idx < K * DIM) {
    int c = idx / K;          // output column of GEMM (0..511)
    int k = idx - c * K;      // reduction index
    WT[idx] = (__bf16)W[k * DIM + c];
  }
}

// --------------------------------------------------------------------------
// Kernel 1: projection GEMM  C[M,512] = A[M,Kdim] @ W + bias, output bf16.
// outMode 0: out[m][c] row-major.  outMode 1 (V): out[b][c][n] (transposed).
// Block: 256 thr (8 waves); wave tile 16(M) x 64(N); grid (M/128, 512/64).
// --------------------------------------------------------------------------
__global__ __launch_bounds__(256) void proj_gemm_kernel(
    const float* __restrict__ A, const __bf16* __restrict__ WT,
    const float* __restrict__ bias, __bf16* __restrict__ out,
    int Kdim, int outMode) {
  const int tid  = threadIdx.x;
  const int wave = tid >> 5;
  const int lane = tid & 31;
  const int l15  = lane & 15;
  const int hi   = lane >> 4;

  const int m0 = blockIdx.x * 128 + wave * 16;
  const int n0 = blockIdx.y * 64;

  v8f acc[4] = {};

  const int row = m0 + l15;                       // A-matrix M per lane
  const float* arow = A + (size_t)row * Kdim;
  const int ksteps = Kdim >> 5;

  for (int kk = 0; kk < ksteps; ++kk) {
    const int kb = (kk << 5) + hi * 16;
    v16bf a = cat16(cvt8(arow + kb), cvt8(arow + kb + 8));
#pragma unroll
    for (int n = 0; n < 4; ++n) {
      const int c = n0 + n * 16 + l15;            // B-matrix N per lane
      v16bf b = *reinterpret_cast<const v16bf*>(
          WT + (size_t)c * Kdim + (kk << 5) + hi * 16);
      acc[n] = WMMA_BF16(a, b, acc[n]);
    }
  }

#pragma unroll
  for (int n = 0; n < 4; ++n) {
    const int c = n0 + n * 16 + l15;
    const float bv = bias[c];
#pragma unroll
    for (int r = 0; r < 8; ++r) {
      const int m = m0 + r + 8 * hi;              // C-layout M = r + 8*hi
      const float v = acc[n][r] + bv;
      if (outMode == 0) {
        out[(size_t)m * DIM + c] = (__bf16)v;
      } else {                                    // V stored as [b][D][N2]
        const int b = m >> 11;
        const int nr = m & 2047;
        out[((size_t)b * DIM + c) * (size_t)N2 + nr] = (__bf16)v;
      }
    }
  }
}

// --------------------------------------------------------------------------
// Kernel 2: flash attention.
// grid (N1/128, B), 256 thr = 8 waves, wave owns 16 q-rows.
// N2 swept in tiles of 64; O accumulator (16x512 f32) lives in LDS per wave.
// LDS: 8 * (16*64 bf16 P-stage + 16*512 f32 O) = 272 KB (dynamic).
// --------------------------------------------------------------------------
__global__ __launch_bounds__(256) void flash_attn_kernel(
    const __bf16* __restrict__ Q, const __bf16* __restrict__ Kmat,
    const __bf16* __restrict__ Vt, float* __restrict__ out) {
  const int tid  = threadIdx.x;
  const int wave = tid >> 5;
  const int lane = tid & 31;
  const int l15  = lane & 15;
  const int hi   = lane >> 4;

  const int b  = blockIdx.y;
  const int q0 = blockIdx.x * 128 + wave * 16;

  const __bf16* Qb  = Q    + ((size_t)b * N1 + q0) * DIM;
  const __bf16* Kb  = Kmat + (size_t)b * N2 * DIM;
  const __bf16* Vtb = Vt   + (size_t)b * DIM * N2;

  extern __shared__ __align__(16) char smem[];
  __bf16* pst = reinterpret_cast<__bf16*>(smem) + wave * (16 * 64);
  float*  osh = reinterpret_cast<float*>(smem + 8 * 16 * 64 * 2) + wave * (16 * DIM);

  // zero wave-private O accumulator (no cross-wave sharing -> no barriers)
  for (int i = lane; i < 16 * DIM; i += 32) osh[i] = 0.0f;

  float mrow[8], lrow[8];
#pragma unroll
  for (int r = 0; r < 8; ++r) { mrow[r] = -3.0e38f; lrow[r] = 0.0f; }

  const float scale = 0.04419417382415922f;  // 1/sqrt(512)
  const __bf16* qrow = Qb + (size_t)l15 * DIM;

  for (int jt = 0; jt < N2 / 64; ++jt) {
    // prefetch next K tile into the cache hierarchy (global_prefetch_b8)
    if (jt + 1 < N2 / 64)
      __builtin_prefetch(Kb + (size_t)((jt + 1) * 64 + lane) * DIM, 0, 3);

    // ---- S = Q K^T (16 x 64 per wave), K-dim = 512 in steps of 32 -------
    v8f s[4] = {};
    for (int kk = 0; kk < DIM / 32; ++kk) {
      const int kb = (kk << 5) + hi * 16;
      v16bf a = cat16(*reinterpret_cast<const v8bf*>(qrow + kb),
                      *reinterpret_cast<const v8bf*>(qrow + kb + 8));
#pragma unroll
      for (int n = 0; n < 4; ++n) {
        const int c = jt * 64 + n * 16 + l15;   // key row per lane
        v16bf kf = *reinterpret_cast<const v16bf*>(
            Kb + (size_t)c * DIM + (kk << 5) + hi * 16);
        s[n] = WMMA_BF16(a, kf, s[n]);
      }
    }

    // ---- online softmax (row stats via 16-lane shfl reductions) ---------
    float alpha[8];
#pragma unroll
    for (int n = 0; n < 4; ++n)
#pragma unroll
      for (int r = 0; r < 8; ++r) s[n][r] *= scale;

#pragma unroll
    for (int r = 0; r < 8; ++r) {
      float rm = s[0][r];
#pragma unroll
      for (int n = 1; n < 4; ++n) rm = fmaxf(rm, s[n][r]);
      rm = redmax16(rm);
      const float nm = fmaxf(mrow[r], rm);
      alpha[r] = __expf(mrow[r] - nm);
      float rs = 0.0f;
#pragma unroll
      for (int n = 0; n < 4; ++n) {
        const float p = __expf(s[n][r] - nm);
        s[n][r] = p;
        rs += p;
      }
      rs = redsum16(rs);
      lrow[r] = lrow[r] * alpha[r] + rs;
      mrow[r] = nm;
    }

    // ---- stage P (bf16) to LDS in plain row-major [16][64] --------------
#pragma unroll
    for (int n = 0; n < 4; ++n)
#pragma unroll
      for (int r = 0; r < 8; ++r)
        pst[(r + 8 * hi) * 64 + n * 16 + l15] = (__bf16)s[n][r];

    // ---- O = O*alpha + P @ V_tile, D swept in chunks of 64 --------------
    for (int dt = 0; dt < DIM / 64; ++dt) {
      v8f o[4] = {};
#pragma unroll
      for (int k2 = 0; k2 < 2; ++k2) {
        const int kb = (k2 << 5) + hi * 16;
        v16bf a = cat16(*reinterpret_cast<const v8bf*>(pst + l15 * 64 + kb),
                        *reinterpret_cast<const v8bf*>(pst + l15 * 64 + kb + 8));
#pragma unroll
        for (int n = 0; n < 4; ++n) {
          const int d = dt * 64 + n * 16 + l15;  // out column per lane
          v16bf vf = *reinterpret_cast<const v16bf*>(
              Vtb + (size_t)d * N2 + jt * 64 + (k2 << 5) + hi * 16);
          o[n] = WMMA_BF16(a, vf, o[n]);
        }
      }
#pragma unroll
      for (int n = 0; n < 4; ++n)
#pragma unroll
        for (int r = 0; r < 8; ++r) {
          const int oi = (r + 8 * hi) * DIM + dt * 64 + n * 16 + l15;
          osh[oi] = osh[oi] * alpha[r] + o[n][r];
        }
    }
  }

  // ---- epilogue: out = O / l ---------------------------------------------
  float* ob = out + ((size_t)b * N1 + q0) * DIM;
#pragma unroll
  for (int r = 0; r < 8; ++r) {
    const int rr = r + 8 * hi;
    const float inv = 1.0f / lrow[r];
    for (int c0 = 0; c0 < DIM; c0 += 16) {
      const int c = c0 + l15;
      ob[(size_t)rr * DIM + c] = osh[rr * DIM + c] * inv;
    }
  }
}

// --------------------------------------------------------------------------
// Host launcher
// --------------------------------------------------------------------------
extern "C" void kernel_launch(void* const* d_in, const int* in_sizes, int n_in,
                              void* d_out, int out_size, void* d_ws, size_t ws_size,
                              hipStream_t stream) {
  const float* input1 = (const float*)d_in[0];  // [B, N1, 512]
  const float* input2 = (const float*)d_in[1];  // [B, N2, 256]
  const float* Wq = (const float*)d_in[2];      // [512, 512]
  const float* bq = (const float*)d_in[3];
  const float* Wk = (const float*)d_in[4];      // [256, 512]
  const float* bk = (const float*)d_in[5];
  const float* Wv = (const float*)d_in[6];      // [256, 512]
  const float* bv = (const float*)d_in[7];

  char* ws = (char*)d_ws;
  size_t off = 0;
  __bf16* Qw  = (__bf16*)(ws + off); off += (size_t)BATCH * N1 * DIM * 2;  // 32 MB
  __bf16* Kw  = (__bf16*)(ws + off); off += (size_t)BATCH * N2 * DIM * 2;  // 32 MB
  __bf16* Vtw = (__bf16*)(ws + off); off += (size_t)BATCH * DIM * N2 * 2;  // 32 MB (transposed V)
  __bf16* WqT = (__bf16*)(ws + off); off += (size_t)DIM * DIM * 2;
  __bf16* WkT = (__bf16*)(ws + off); off += (size_t)DIM * LEN * 2;
  __bf16* WvT = (__bf16*)(ws + off); off += (size_t)DIM * LEN * 2;

  // 0) transpose + bf16-convert weights
  wt_transpose_kernel<<<(DIM * DIM + 255) / 256, 256, 0, stream>>>(Wq, WqT, DIM);
  wt_transpose_kernel<<<(DIM * LEN + 255) / 256, 256, 0, stream>>>(Wk, WkT, LEN);
  wt_transpose_kernel<<<(DIM * LEN + 255) / 256, 256, 0, stream>>>(Wv, WvT, LEN);

  // 1) projections (M = B*2048 = 32768 rows)
  dim3 pgrid(32768 / 128, DIM / 64);
  proj_gemm_kernel<<<pgrid, 256, 0, stream>>>(input1, WqT, bq, Qw, DIM, 0);
  proj_gemm_kernel<<<pgrid, 256, 0, stream>>>(input2, WkT, bk, Kw, LEN, 0);
  proj_gemm_kernel<<<pgrid, 256, 0, stream>>>(input2, WvT, bv, Vtw, LEN, 1);

  // 2) fused attention; 272 KB dynamic LDS per workgroup (one WGP each)
  const size_t smem = 8 * (16 * 64 * 2 + 16 * DIM * 4);
  flash_attn_kernel<<<dim3(N1 / 128, BATCH), 256, smem, stream>>>(
      Qw, Kw, Vtw, (float*)d_out);
}
